// MultimodalSynchronyTracker_17239998726203
// MI455X (gfx1250) — matrix operand order repaired and checked by reading, not compiled, not run on hardware
//
#include <hip/hip_runtime.h>
#include <hip/hip_bf16.h>

// ---------------------------------------------------------------------------
// MultimodalSynchronyTracker forward for MI455X (gfx1250, wave32, WMMA).
// All GEMMs: v_wmma_f32_16x16x32_bf16, bf16 inputs (pre-converted weights,
// register-hoisted A tiles), f32 accumulate.
// ---------------------------------------------------------------------------

typedef __attribute__((ext_vector_type(16))) __bf16 v16bf;
typedef __attribute__((ext_vector_type(8)))  float  v8f;

// ---- model dims -----------------------------------------------------------
constexpr int Bb = 128, NCt = 768, NPt = 640, DMc = 256, DSY = 64;
constexpr int NALIGN = 256, WINc = 16, STRIDEc = 8;
constexpr int NWIN = (NALIGN - WINc) / STRIDEc + 1;       // 31
constexpr int DSc = 16, DIc = 512, DTRc = 16;
constexpr int M_CT = Bb * NCt;                            // 98304 rows
constexpr int M_CL = Bb * NPt;                            // 81920 rows
constexpr int M_H  = Bb * NWIN;                           // 3968 rows

// ---- workspace layout (bytes) --------------------------------------------
constexpr size_t alup(size_t x) { return (x + 255) & ~(size_t)255; }
constexpr size_t OFF_BPC = 0;                                        // f32 98304x64
constexpr size_t OFF_BPP = OFF_BPC + alup((size_t)M_CT * DSY * 4);   // f32 81920x64
constexpr size_t OFF_H1  = OFF_BPP + alup((size_t)M_CL * DSY * 4);   // bf16 98304x64
constexpr size_t OFF_H   = OFF_H1;                                   // f32 3968x256 (h1 dead)
// Mamba scratch reuses the bp region [0, OFF_H1):
constexpr size_t OFF_LN  = 0;                                        // bf16 3968x256
constexpr size_t OFF_XZ  = OFF_LN  + alup((size_t)M_H * DMc * 2);    // f32 3968x1024
constexpr size_t OFF_XCF = OFF_XZ  + alup((size_t)M_H * 1024 * 4);   // f32 3968x512
constexpr size_t OFF_XCB = OFF_XCF + alup((size_t)M_H * DIc * 4);    // bf16 3968x512
constexpr size_t OFF_DBL = OFF_XCB + alup((size_t)M_H * DIc * 2);    // f32 3968x48
constexpr size_t OFF_DT  = OFF_DBL + alup((size_t)M_H * 48 * 4);     // f32 3968x512
constexpr size_t OFF_YA  = OFF_DT  + alup((size_t)M_H * DIc * 4);    // bf16 3968x512
constexpr size_t SCRATCH_END = OFF_YA + alup((size_t)M_H * DIc * 2);
static_assert(SCRATCH_END <= OFF_H1, "mamba scratch must not clobber h");

// bf16 weight cache region: lives after the h1 region, never aliased.
constexpr size_t OFF_WB  = OFF_H1 + alup((size_t)M_CT * DSY * 2);
constexpr size_t WB_BPW1 = 0;                       // 64x256
constexpr size_t WB_BPW2 = WB_BPW1 + 64 * 256;      // 64x64
constexpr size_t WB_L0   = WB_BPW2 + 64 * 64;
constexpr size_t WB_IN   = 0;                       // per-layer: 1024x256
constexpr size_t WB_XP   = WB_IN + 1024 * 256;      // 48x512
constexpr size_t WB_DT   = WB_XP + 48 * 512;        // 512x32 (zero-padded K)
constexpr size_t WB_OUT  = WB_DT + 512 * 32;        // 256x512
constexpr size_t WB_LSZ  = WB_OUT + 256 * 512;      // 434176 elems/layer

// ---- helpers --------------------------------------------------------------
__device__ __forceinline__ unsigned short f32_to_bf16(float f) {
  unsigned u = __builtin_bit_cast(unsigned, f);
  u += 0x7FFFu + ((u >> 16) & 1u);               // round-to-nearest-even
  return (unsigned short)(u >> 16);
}
__device__ __forceinline__ __bf16 bfraw(unsigned short h) {
  return __builtin_bit_cast(__bf16, h);
}

enum { ACT_NONE = 0, ACT_GELU = 1, ACT_SOFTPLUS = 2 };

// A tile (16x32 slice) from f32 row-major; ISA 16-bit A layout:
// lane 0-15: M=lane, K halves {0..7,16..23}; lane 16-31: {8..15,24..31}.
__device__ __forceinline__ v16bf load_a_f32(const float* A, int lda, int m0,
                                            int kbase, int lane) {
  int row  = m0 + (lane & 15);
  int koff = kbase + ((lane >> 4) << 3);
  const float* p = A + (size_t)row * lda + koff;
  v16bf a;
#pragma unroll
  for (int e = 0; e < 8; ++e) {
    a[e]     = bfraw(f32_to_bf16(p[e]));
    a[8 + e] = bfraw(f32_to_bf16(p[16 + e]));
  }
  return a;
}

__device__ __forceinline__ v16bf load_a_bf16(const unsigned short* A, int lda,
                                             int m0, int kbase, int lane) {
  int row  = m0 + (lane & 15);
  int koff = kbase + ((lane >> 4) << 3);
  const unsigned short* p = A + (size_t)row * lda + koff;
  v16bf a;
#pragma unroll
  for (int e = 0; e < 8; ++e) { a[e] = bfraw(p[e]); a[8 + e] = bfraw(p[16 + e]); }
  return a;
}

// B tile (32x16) from pre-converted bf16 weight W (N,K) row-major (X @ W^T).
// ISA 16-bit B layout: lane = N col; lanes 0-15 hold K {0..7,16..23}, 16-31 hold {8..15,24..31}.
__device__ __forceinline__ v16bf load_b_bf16(const unsigned short* W, int ldw,
                                             int n0, int kbase, int lane) {
  int n  = n0 + (lane & 15);
  int kb = kbase + ((lane >> 4) << 3);
  const unsigned short* p = W + (size_t)n * ldw + kb;
  v16bf b;
#pragma unroll
  for (int e = 0; e < 8; ++e) { b[e] = bfraw(p[e]); b[8 + e] = bfraw(p[16 + e]); }
  return b;
}

// ---- weight f32 -> bf16 pre-conversion (with optional K zero-padding) ----
__global__ __launch_bounds__(256) void cvt_w_bf16_kernel(
    const float* __restrict__ src, unsigned short* __restrict__ dst,
    int N, int Kin, int Kout) {
  int i = blockIdx.x * blockDim.x + threadIdx.x;
  if (i >= N * Kout) return;
  int n = i / Kout, k = i - n * Kout;
  dst[i] = (k < Kin) ? f32_to_bf16(src[(size_t)n * Kin + k]) : (unsigned short)0;
}

// ---- generic WMMA GEMM: Y = act(A @ W^T + bias) [+ Y if ACCUM] ------------
// A tiles for the whole K extent are hoisted into registers and reused
// across all column tiles handled by this wave.
template<int KP, int ACT, bool A_BF16, bool ACCUM, bool ST_F32, bool ST_BF16>
__global__ __launch_bounds__(128) void gemm_wmma_kernel(
    const void* __restrict__ Ap, int lda,
    const unsigned short* __restrict__ Wb, int ldw,
    const float* __restrict__ bias,
    float* __restrict__ Yf, unsigned short* __restrict__ Ybf, int ldy, int N) {
  constexpr int KT = KP / 32;
  const int wave = threadIdx.x >> 5;
  const int lane = threadIdx.x & 31;
  const int m0   = (blockIdx.x * 4 + wave) * 16;

  v16bf at[KT];
#pragma unroll
  for (int kt = 0; kt < KT; ++kt) {
    if constexpr (A_BF16)
      at[kt] = load_a_bf16((const unsigned short*)Ap, lda, m0, kt * 32, lane);
    else
      at[kt] = load_a_f32((const float*)Ap, lda, m0, kt * 32, lane);
  }

  const int yhi  = ((int)blockIdx.y + 1) * 64;
  const int nend = (N < yhi) ? N : yhi;
  for (int n0 = (int)blockIdx.y * 64; n0 < nend; n0 += 16) {
    if (n0 + 16 < nend)   // gfx1250 global_prefetch_b8 for next column tile
      __builtin_prefetch(Wb + (size_t)(n0 + 16 + (lane & 15)) * ldw +
                         ((lane >> 4) << 4), 0, 1);
    const int col   = n0 + (lane & 15);
    const int rbase = m0 + ((lane >> 4) << 3);
    v8f c;
    if constexpr (ACCUM) {
#pragma unroll
      for (int r = 0; r < 8; ++r) c[r] = Yf[(size_t)(rbase + r) * ldy + col];
    } else {
      float bv = bias ? bias[col] : 0.f;
#pragma unroll
      for (int r = 0; r < 8; ++r) c[r] = bv;
    }
#pragma unroll
    for (int kt = 0; kt < KT; ++kt) {
      v16bf b = load_b_bf16(Wb, ldw, n0, kt * 32, lane);
      c = __builtin_amdgcn_wmma_f32_16x16x32_bf16(false, at[kt], false, b,
                                                  (short)0, c, false, false);
    }
#pragma unroll
    for (int r = 0; r < 8; ++r) {
      float v = c[r];
      if constexpr (ACT == ACT_GELU)     v = 0.5f * v * (1.f + erff(v * 0.70710678f));
      if constexpr (ACT == ACT_SOFTPLUS) v = (v > 20.f) ? v : log1pf(expf(v));
      size_t o = (size_t)(rbase + r) * ldy + col;
      if constexpr (ST_F32)  Yf[o]  = v;
      if constexpr (ST_BF16) Ybf[o] = f32_to_bf16(v);
    }
  }
}

// ---- fused temporal-resample + window stats + 3->256 projection ----------
__global__ __launch_bounds__(128) void sync_stats_kernel(
    const float* __restrict__ bp_c, const float* __restrict__ bp_p,
    const float* __restrict__ proj_w, const float* __restrict__ proj_b,
    float* __restrict__ h) {
  const int wave = threadIdx.x >> 5, lane = threadIdx.x & 31;
  const int wid = blockIdx.x * 4 + wave;              // 0..3967
  const int b = wid / NWIN, w = wid % NWIN;
  float st = 0, sc = 0, stt = 0, scc = 0, stc = 0;
#pragma unroll 4
  for (int i = 0; i < 32; ++i) {
    int e = lane + 32 * i;                            // 0..1023 within window
    int tl = e >> 6, d = e & 63;
    int t = w * STRIDEc + tl;
    // counselor: resample 768 -> 256 (ratio 3)
    float s0 = fmaxf((t + 0.5f) * 3.0f - 0.5f, 0.f);
    int lo0 = (int)floorf(s0);
    int hi0 = (lo0 + 1 < NCt - 1) ? lo0 + 1 : NCt - 1;
    float fr0 = s0 - (float)lo0;
    const float* pc = bp_c + (size_t)b * NCt * DSY + d;
    float tv = pc[(size_t)lo0 * DSY] * (1.f - fr0) + pc[(size_t)hi0 * DSY] * fr0;
    // client: resample 640 -> 256 (ratio 2.5)
    float s1 = fmaxf((t + 0.5f) * 2.5f - 0.5f, 0.f);
    int lo1 = (int)floorf(s1);
    int hi1 = (lo1 + 1 < NPt - 1) ? lo1 + 1 : NPt - 1;
    float fr1 = s1 - (float)lo1;
    const float* pp = bp_p + (size_t)b * NPt * DSY + d;
    float cv = pp[(size_t)lo1 * DSY] * (1.f - fr1) + pp[(size_t)hi1 * DSY] * fr1;
    st += tv; sc += cv; stt += tv * tv; scc += cv * cv; stc += tv * cv;
  }
#pragma unroll
  for (int off = 16; off; off >>= 1) {
    st  += __shfl_xor(st, off, 32);  sc  += __shfl_xor(sc, off, 32);
    stt += __shfl_xor(stt, off, 32); scc += __shfl_xor(scc, off, 32);
    stc += __shfl_xor(stc, off, 32);
  }
  const float n = (float)(WINc * DSY), EPS = 1e-8f;
  float tn = fmaxf(sqrtf(stt), EPS), cn = fmaxf(sqrtf(scc), EPS);
  float cosv = stc / (tn * cn);
  float mt = st / n, mc = sc / n;
  float cov = stc - n * mt * mc;
  float vt = fmaxf(stt - n * mt * mt, 0.f), vc = fmaxf(scc - n * mc * mc, 0.f);
  float pear = cov / (fmaxf(sqrtf(vt), EPS) * fmaxf(sqrtf(vc), EPS));
  float l2 = sqrtf(fmaxf(stt - 2.f * stc + scc, 0.f));
  float* hp = h + (size_t)wid * DMc;
#pragma unroll
  for (int i = 0; i < 8; ++i) {
    int d = lane + 32 * i;
    hp[d] = proj_w[d * 3] * cosv + proj_w[d * 3 + 1] * pear +
            proj_w[d * 3 + 2] * l2 + proj_b[d];
  }
}

// ---- LayerNorm over D=256 (one wave per row) ------------------------------
template<bool TO_BF16>
__global__ __launch_bounds__(128) void layernorm_kernel(
    const float* __restrict__ X, const float* __restrict__ g,
    const float* __restrict__ bta, float* __restrict__ Yf,
    unsigned short* __restrict__ Ybf) {
  const int wave = threadIdx.x >> 5, lane = threadIdx.x & 31;
  const int row = blockIdx.x * 4 + wave;
  const float* x = X + (size_t)row * DMc;
  float vals[8]; float s = 0, s2 = 0;
#pragma unroll
  for (int i = 0; i < 8; ++i) {
    float v = x[lane + 32 * i]; vals[i] = v; s += v; s2 += v * v;
  }
#pragma unroll
  for (int off = 16; off; off >>= 1) {
    s += __shfl_xor(s, off, 32); s2 += __shfl_xor(s2, off, 32);
  }
  float mean = s * (1.f / DMc);
  float var = s2 * (1.f / DMc) - mean * mean;
  float inv = rsqrtf(var + 1e-5f);
#pragma unroll
  for (int i = 0; i < 8; ++i) {
    int d = lane + 32 * i;
    float y = (vals[i] - mean) * inv * g[d] + bta[d];
    size_t o = (size_t)row * DMc + d;
    if constexpr (TO_BF16) Ybf[o] = f32_to_bf16(y);
    else                   Yf[o]  = y;
  }
}

// ---- depthwise causal conv (width 4) + SiLU -------------------------------
__global__ __launch_bounds__(256) void conv_silu_kernel(
    const float* __restrict__ xz, const float* __restrict__ cw,
    const float* __restrict__ cb, float* __restrict__ xcf,
    unsigned short* __restrict__ xcbf) {
  int idx = blockIdx.x * blockDim.x + threadIdx.x;     // b*512 + ch
  int b = idx >> 9, ch = idx & 511;
  float w0 = cw[ch * 4], w1 = cw[ch * 4 + 1], w2 = cw[ch * 4 + 2], w3 = cw[ch * 4 + 3];
  float bias = cb[ch];
  float x0 = 0.f, x1 = 0.f, x2 = 0.f;
  const float* src = xz + (size_t)(b * NWIN) * 1024 + ch;
  for (int t = 0; t < NWIN; ++t) {
    float x3 = src[(size_t)t * 1024];
    float v = x0 * w0 + x1 * w1 + x2 * w2 + x3 * w3 + bias;
    v = v / (1.f + expf(-v));                          // SiLU
    size_t o = (size_t)(b * NWIN + t) * DIc + ch;
    xcf[o] = v; xcbf[o] = f32_to_bf16(v);
    x0 = x1; x1 = x2; x2 = x3;
  }
}

// ---- selective scan (state DS=16 per channel) + D skip + z-gate -----------
__global__ __launch_bounds__(512) void scan_kernel(
    const float* __restrict__ dtb, const float* __restrict__ dbl,
    const float* __restrict__ xcf, const float* __restrict__ xz,
    const float* __restrict__ A_log, const float* __restrict__ Dp,
    unsigned short* __restrict__ yact) {
  const int b = blockIdx.x, ch = threadIdx.x;          // 512 channels
  float A[16];
#pragma unroll
  for (int s = 0; s < 16; ++s) A[s] = -expf(A_log[ch * 16 + s]);
  const float Dv = Dp[ch];
  float hs[16];
#pragma unroll
  for (int s = 0; s < 16; ++s) hs[s] = 0.f;
  for (int t = 0; t < NWIN; ++t) {
    const int r = b * NWIN + t;
    float dt = dtb[(size_t)r * DIc + ch];
    float x  = xcf[(size_t)r * DIc + ch];
    float z  = xz[(size_t)r * 1024 + DIc + ch];
    const float* Bm = dbl + (size_t)r * 48 + DTRc;       // [16..32)
    const float* Cm = dbl + (size_t)r * 48 + DTRc + DSc; // [32..48)
    float y = 0.f;
#pragma unroll
    for (int s = 0; s < 16; ++s) {
      float dA = expf(dt * A[s]);
      hs[s] = dA * hs[s] + dt * Bm[s] * x;
      y += hs[s] * Cm[s];
    }
    y += Dv * x;
    float sz = z / (1.f + expf(-z));
    yact[(size_t)r * DIc + ch] = f32_to_bf16(y * sz);
  }
}

// ---------------------------------------------------------------------------
extern "C" void kernel_launch(void* const* d_in, const int* in_sizes, int n_in,
                              void* d_out, int out_size, void* d_ws, size_t ws_size,
                              hipStream_t stream) {
  (void)n_in; (void)out_size; (void)ws_size;
  auto F = [&](int i) { return (const float*)d_in[i]; };

  const float *counselor, *client;
  const float *bp_w1, *bp_b1, *bp_w2, *bp_b2, *proj_w, *proj_b, *fn_g, *fn_b;
  struct LP { const float *ln_g, *ln_b, *in_w, *conv_w, *conv_b, *xproj_w,
              *dt_w, *dt_b, *A_log, *D, *out_w; } L[2];

  if (in_sizes[0] == Bb * NCt * DMc) {
    // insertion-order flattening
    counselor = F(0); client = F(1);
    bp_w1 = F(2); bp_b1 = F(3); bp_w2 = F(4); bp_b2 = F(5);
    proj_w = F(6); proj_b = F(7);
    for (int l = 0; l < 2; ++l) {
      int k = 8 + l * 11;
      L[l] = { F(k), F(k+1), F(k+2), F(k+3), F(k+4), F(k+5),
               F(k+6), F(k+7), F(k+8), F(k+9), F(k+10) };
    }
    fn_g = F(30); fn_b = F(31);
  } else {
    // JAX tree_leaves (sorted-key) flattening
    client = F(0); counselor = F(1);
    bp_b1 = F(2); bp_b2 = F(3); bp_w1 = F(4); bp_w2 = F(5);
    fn_b = F(6); fn_g = F(7);
    for (int l = 0; l < 2; ++l) {
      int k = 8 + l * 11;
      // sorted: A_log, D, conv_b, conv_w, dt_b, dt_w, in_w, ln_b, ln_g, out_w, xproj_w
      L[l].A_log = F(k);     L[l].D = F(k+1);    L[l].conv_b = F(k+2);
      L[l].conv_w = F(k+3);  L[l].dt_b = F(k+4); L[l].dt_w = F(k+5);
      L[l].in_w = F(k+6);    L[l].ln_b = F(k+7); L[l].ln_g = F(k+8);
      L[l].out_w = F(k+9);   L[l].xproj_w = F(k+10);
    }
    proj_b = F(30); proj_w = F(31);
  }

  char* ws = (char*)d_ws;
  float*          bp_c = (float*)(ws + OFF_BPC);
  float*          bp_p = (float*)(ws + OFF_BPP);
  unsigned short* h1   = (unsigned short*)(ws + OFF_H1);
  float*          hbuf = (float*)(ws + OFF_H);
  unsigned short* lnb  = (unsigned short*)(ws + OFF_LN);
  float*          xz   = (float*)(ws + OFF_XZ);
  float*          xcf  = (float*)(ws + OFF_XCF);
  unsigned short* xcb  = (unsigned short*)(ws + OFF_XCB);
  float*          dbl  = (float*)(ws + OFF_DBL);
  float*          dtb  = (float*)(ws + OFF_DT);
  unsigned short* yab  = (unsigned short*)(ws + OFF_YA);
  unsigned short* wb   = (unsigned short*)(ws + OFF_WB);

  unsigned short* bp_w1b = wb + WB_BPW1;
  unsigned short* bp_w2b = wb + WB_BPW2;
  unsigned short *in_wb[2], *xp_wb[2], *dt_wb[2], *out_wb[2];
  for (int l = 0; l < 2; ++l) {
    unsigned short* base = wb + WB_L0 + (size_t)l * WB_LSZ;
    in_wb[l]  = base + WB_IN;
    xp_wb[l]  = base + WB_XP;
    dt_wb[l]  = base + WB_DT;
    out_wb[l] = base + WB_OUT;
  }

  // ---- pre-convert weights to bf16 (dt_w zero-padded K 16 -> 32) ---------
  auto cvt = [&](const float* s, unsigned short* d, int N, int Kin, int Kout) {
    int cnt = N * Kout;
    cvt_w_bf16_kernel<<<(cnt + 255) / 256, 256, 0, stream>>>(s, d, N, Kin, Kout);
  };
  cvt(bp_w1, bp_w1b, 64, 256, 256);
  cvt(bp_w2, bp_w2b, 64, 64, 64);
  for (int l = 0; l < 2; ++l) {
    cvt(L[l].in_w,    in_wb[l],  1024, 256, 256);
    cvt(L[l].xproj_w, xp_wb[l],  48,   512, 512);
    cvt(L[l].dt_w,    dt_wb[l],  512,  16,  32);    // zero-pad K
    cvt(L[l].out_w,   out_wb[l], 256,  512, 512);
  }

  // ---- behavior_proj: counselor (M=98304) then client (M=81920) ----------
  gemm_wmma_kernel<256, ACT_GELU, false, false, false, true>
      <<<dim3(M_CT / 64, 1), 128, 0, stream>>>(counselor, DMc,
          bp_w1b, DMc, bp_b1, nullptr, h1, DSY, DSY);
  gemm_wmma_kernel<64, ACT_NONE, true, false, true, false>
      <<<dim3(M_CT / 64, 1), 128, 0, stream>>>(h1, DSY,
          bp_w2b, DSY, bp_b2, bp_c, nullptr, DSY, DSY);
  gemm_wmma_kernel<256, ACT_GELU, false, false, false, true>
      <<<dim3(M_CL / 64, 1), 128, 0, stream>>>(client, DMc,
          bp_w1b, DMc, bp_b1, nullptr, h1, DSY, DSY);
  gemm_wmma_kernel<64, ACT_NONE, true, false, true, false>
      <<<dim3(M_CL / 64, 1), 128, 0, stream>>>(h1, DSY,
          bp_w2b, DSY, bp_b2, bp_p, nullptr, DSY, DSY);

  // ---- fused resample + window stats + 3->256 projection -> h ------------
  sync_stats_kernel<<<M_H / 4, 128, 0, stream>>>(bp_c, bp_p, proj_w, proj_b, hbuf);

  // ---- two Mamba blocks ---------------------------------------------------
  for (int l = 0; l < 2; ++l) {
    layernorm_kernel<true><<<M_H / 4, 128, 0, stream>>>(
        hbuf, L[l].ln_g, L[l].ln_b, nullptr, lnb);
    // xz = ln @ in_w^T   (K=256 -> N=1024)
    gemm_wmma_kernel<256, ACT_NONE, true, false, true, false>
        <<<dim3(M_H / 64, 16), 128, 0, stream>>>(lnb, DMc,
            in_wb[l], DMc, nullptr, xz, nullptr, 1024, 1024);
    conv_silu_kernel<<<(Bb * DIc) / 256, 256, 0, stream>>>(
        xz, L[l].conv_w, L[l].conv_b, xcf, xcb);
    // dbl = xc @ xproj_w^T (K=512 -> N=48)
    gemm_wmma_kernel<512, ACT_NONE, true, false, true, false>
        <<<dim3(M_H / 64, 1), 128, 0, stream>>>(xcb, DIc,
            xp_wb[l], DIc, nullptr, dbl, nullptr, 48, 48);
    // dt = softplus(dbl[:, :16] @ dt_w^T + dt_b); K padded to 32, weight rows
    // k>=16 are zero so the A-side over-read of dbl contributes nothing.
    gemm_wmma_kernel<32, ACT_SOFTPLUS, false, false, true, false>
        <<<dim3(M_H / 64, 8), 128, 0, stream>>>(dbl, 48,
            dt_wb[l], 32, L[l].dt_b, dtb, nullptr, DIc, DIc);
    scan_kernel<<<Bb, DIc, 0, stream>>>(dtb, dbl, xcf, xz,
                                        L[l].A_log, L[l].D, yab);
    // h += y_act @ out_w^T (K=512 -> N=256, residual via WMMA C operand)
    gemm_wmma_kernel<512, ACT_NONE, true, true, true, false>
        <<<dim3(M_H / 64, 4), 128, 0, stream>>>(yab, DIc,
            out_wb[l], DIc, nullptr, hbuf, nullptr, DMc, DMc);
  }

  // ---- final LayerNorm -> d_out (f32) -------------------------------------
  layernorm_kernel<false><<<M_H / 4, 128, 0, stream>>>(
      hbuf, fn_g, fn_b, (float*)d_out, nullptr);
}
